// TypedHGNN_41566693491334
// MI455X (gfx1250) — matrix operand order repaired and checked by reading, not compiled, not run on hardware
//
#include <hip/hip_runtime.h>
#include <hip/hip_bf16.h>
#include <math.h>

// ---------------------------------------------------------------- constants
#define NN   4096   // nodes
#define EE   2048   // hyperedges
#define RR   4      // relations
#define FF   128    // in/out features
#define HD   256    // hidden dim
#define DEG_EPS 1e-8f
#define BN_EPS  1e-5f
#define LN_EPS  1e-5f

typedef __attribute__((ext_vector_type(16))) __bf16 v16bf;
typedef __attribute__((ext_vector_type(8)))  float  v8f;
typedef __attribute__((ext_vector_type(4)))  int    v4i;

// CDNA5 async global->LDS (ASYNCcnt) if the toolchain exposes it
#if defined(__HIP_DEVICE_COMPILE__) && __has_builtin(__builtin_amdgcn_global_load_async_to_lds_b128) && __has_builtin(__builtin_amdgcn_s_wait_asynccnt)
#define HAVE_ASYNC 1
typedef __attribute__((address_space(1))) v4i* v4i_glob;
typedef __attribute__((address_space(3))) v4i* v4i_lds;
#else
#define HAVE_ASYNC 0
#endif

// ---------------------------------------------------------------- helpers
__device__ __forceinline__ float sigmoidf_(float x) { return 1.0f / (1.0f + __expf(-x)); }

// ---------------------------------------------------------------- prep: softmax over rel_attn (R=4)
__global__ void k_prep(const float* __restrict__ ra, float* __restrict__ rw) {
    if (threadIdx.x == 0) {
        float m = ra[0];
        for (int i = 1; i < RR; ++i) m = fmaxf(m, ra[i]);
        float e[RR], s = 0.f;
        for (int i = 0; i < RR; ++i) { e[i] = __expf(ra[i] - m); s += e[i]; }
        for (int i = 0; i < RR; ++i) rw[i] = e[i] / s;
    }
}

// ---------------------------------------------------------------- node scale: rsqrt(rw*rowsum+eps), one block per (r,n)
__global__ __launch_bounds__(256) void k_scaleN(const float* __restrict__ H,
                                                const float* __restrict__ rw,
                                                float* __restrict__ sN) {
    __shared__ float red[256];
    int n = blockIdx.x, r = blockIdx.y, tid = threadIdx.x;
    const float* row = H + ((long)r * NN + n) * EE;
    float s = 0.f;
    for (int e = tid; e < EE; e += 256) s += row[e];
    red[tid] = s; __syncthreads();
    for (int w = 128; w > 0; w >>= 1) { if (tid < w) red[tid] += red[tid + w]; __syncthreads(); }
    if (tid == 0) sN[(long)r * NN + n] = rsqrtf(rw[r] * red[0] + DEG_EPS);
}

// ---------------------------------------------------------------- edge scale: columns, coalesced
__global__ __launch_bounds__(256) void k_scaleE(const float* __restrict__ H,
                                                const float* __restrict__ rw,
                                                float* __restrict__ sE) {
    int idx = blockIdx.x * 256 + threadIdx.x;       // over R*EE
    int r = idx / EE, e = idx % EE;
    const float* base = H + (long)r * NN * EE + e;
    float s = 0.f;
    for (int n = 0; n < NN; ++n) s += base[(long)n * EE];
    sE[idx] = rsqrtf(rw[r] * s + DEG_EPS);
}

// ---------------------------------------------------------------- theta = H * rw * sN * sE  (bf16, L2-resident afterwards)
__global__ __launch_bounds__(256) void k_theta(const float* __restrict__ H,
                                               const float* __restrict__ rw,
                                               const float* __restrict__ sN,
                                               const float* __restrict__ sE,
                                               __bf16* __restrict__ theta) {
    long idx = (long)blockIdx.x * 256 + threadIdx.x;   // over R*NN*EE
    int e = (int)(idx % EE);
    long t = idx / EE;
    int n = (int)(t % NN);
    int r = (int)(t / NN);
    float v = H[idx] * rw[r] * sN[(long)r * NN + n] * sE[(long)r * EE + e];
    theta[idx] = (__bf16)v;
}

// ---------------------------------------------------------------- f32 -> bf16 convert
__global__ __launch_bounds__(256) void k_cvt(const float* __restrict__ src, __bf16* __restrict__ dst) {
    long i = (long)blockIdx.x * 256 + threadIdx.x;
    dst[i] = (__bf16)src[i];
}

// ---------------------------------------------------------------- weights * sigmoid(imp) -> bf16
__global__ __launch_bounds__(256) void k_wconv(const float* __restrict__ W, const float* __restrict__ imp,
                                               int layer, int per_rel, __bf16* __restrict__ Wb) {
    long i = (long)blockIdx.x * 256 + threadIdx.x;   // over RR*per_rel
    int r = (int)(i / per_rel);
    Wb[i] = (__bf16)(W[i] * sigmoidf_(imp[layer * RR + r]));
}

// ================================================================ bf16 WMMA GEMM
// C[M x Ncol] (f32) = A[M x K] * B[K x Ncol], operands bf16.
// TRANSA: A stored as [K x M] row-major (element (m,k) at A[k*ldA + m]).
// REDUCE: accumulate over nbatch relations into one C, epilogue adds bias (+resid).
// Block: 256 threads (8 wave32), tile BM=128 (16 rows/wave), BN=64 (4 wmma/wave), BK=32.
// LDS: A tile row-major [m][k] (stride 40), B tile TRANSPOSED [n][k] (stride 40)
//      so every per-lane WMMA fragment is two contiguous 16-byte chunks -> ds_load_b128.
template<bool TRANSA, bool REDUCE>
__global__ __launch_bounds__(256) void gemm_bf16(
    const __bf16* __restrict__ A, const __bf16* __restrict__ B, float* __restrict__ C,
    int M, int Ncol, int K, int ldA, int ldB, int ldC,
    long sA, long sB, long sC, int nbatch,
    const float* __restrict__ bias, const float* __restrict__ resid) {

    __shared__ __align__(16) __bf16 As[128 * 40];   // [m][k], k-stride 40 (80B rows, 16B aligned)
    __shared__ __align__(16) __bf16 Bt[64 * 40];    // [n][k], transposed B tile

    const int tid  = threadIdx.x;
    const int wave = tid >> 5;
    const int lane = tid & 31;
    const int m0 = blockIdx.y * 128;
    const int nb = blockIdx.x * 64;

    v8f acc[4];
    #pragma unroll
    for (int j = 0; j < 4; ++j)
        #pragma unroll
        for (int i = 0; i < 8; ++i) acc[j][i] = 0.f;

    const int rb = REDUCE ? 0 : blockIdx.z;
    const int re = REDUCE ? nbatch : blockIdx.z + 1;

    const int mm   = lane & 15;
    const int half = lane >> 4;
    const int w16  = wave * 16;

    union Frag  { uint4 q[2]; v16bf v; };
    union Ld16  { uint4 q[2]; __bf16 h[16]; };
    union Ld8   { uint4 q;    __bf16 h[8]; };

    for (int r = rb; r < re; ++r) {
        const __bf16* Ar = A + (long)r * sA;
        const __bf16* Br = B + (long)r * sB;
        for (int kb = 0; kb < K; kb += 32) {
            __syncthreads();
            // ---- stage A tile into [m][k]
            if (TRANSA) {
                // global rows are along m (row k holds contiguous m) -> scatter-transpose
                int kr = tid >> 3, ch = tid & 7;
                Ld16 t;
                const uint4* s = reinterpret_cast<const uint4*>(Ar + (long)(kb + kr) * ldA + m0 + ch * 16);
                t.q[0] = s[0]; t.q[1] = s[1];
                #pragma unroll
                for (int jj = 0; jj < 16; ++jj) As[(ch * 16 + jj) * 40 + kr] = t.h[jj];
            } else {
                int row = tid >> 1, hf = tid & 1;
                const __bf16* src = Ar + (long)(m0 + row) * ldA + kb + hf * 16;
                __bf16* dst = &As[row * 40 + hf * 16];
#if HAVE_ASYNC
                __bf16* srcnc = const_cast<__bf16*>(src);
                __builtin_amdgcn_global_load_async_to_lds_b128(
                    (v4i_glob)srcnc, (v4i_lds)dst, 0, 0);
                __builtin_amdgcn_global_load_async_to_lds_b128(
                    (v4i_glob)(srcnc + 8), (v4i_lds)(dst + 8), 0, 0);
#else
                const uint4* s = reinterpret_cast<const uint4*>(src);
                uint4* d = reinterpret_cast<uint4*>(dst);
                d[0] = s[0]; d[1] = s[1];
#endif
            }
            // ---- stage B tile transposed into [n][k]
            {
                int kr = tid >> 3, ch = tid & 7;
                Ld8 t;
                t.q = *reinterpret_cast<const uint4*>(Br + (long)(kb + kr) * ldB + nb + ch * 8);
                #pragma unroll
                for (int jj = 0; jj < 8; ++jj) Bt[(ch * 8 + jj) * 40 + kr] = t.h[jj];
            }
            // prefetch next K tile into cache (global_prefetch_b8)
            if (kb + 32 < K) {
                __builtin_prefetch(Ar + (long)(TRANSA ? (kb + 32) * ldA + m0 : m0 * ldA + kb + 32));
                __builtin_prefetch(Br + (long)(kb + 32) * ldB + nb);
            }
#if HAVE_ASYNC
            if (!TRANSA) __builtin_amdgcn_s_wait_asynccnt(0);
#endif
            __syncthreads();

            // ---- A fragment + all 4 B fragments up front (one dscnt wait covers all),
            //      then 4 back-to-back WMMAs on disjoint registers
            Frag fa;
            fa.q[0] = *reinterpret_cast<const uint4*>(&As[(w16 + mm) * 40 + 8 * half]);
            fa.q[1] = *reinterpret_cast<const uint4*>(&As[(w16 + mm) * 40 + 16 + 8 * half]);

            Frag fb[4];
            #pragma unroll
            for (int j = 0; j < 4; ++j) {
                int n = j * 16 + mm;
                fb[j].q[0] = *reinterpret_cast<const uint4*>(&Bt[n * 40 + 8 * half]);
                fb[j].q[1] = *reinterpret_cast<const uint4*>(&Bt[n * 40 + 16 + 8 * half]);
            }
            #pragma unroll
            for (int j = 0; j < 4; ++j) {
                acc[j] = __builtin_amdgcn_wmma_f32_16x16x32_bf16(
                    false, fa.v, false, fb[j].v, (short)0, acc[j], false, false);
            }
        }
    }

    // ---- epilogue: C layout lanes0-15 M=i, lanes16-31 M=i+8
    #pragma unroll
    for (int j = 0; j < 4; ++j) {
        #pragma unroll
        for (int i = 0; i < 8; ++i) {
            long row = m0 + w16 + i + 8 * half;
            long col = nb + j * 16 + mm;
            float v = acc[j][i];
            if (REDUCE) {
                if (bias)  v += bias[col];
                if (resid) v += resid[row * ldC + col];
                C[row * ldC + col] = v;
            } else {
                C[(long)blockIdx.z * sC + row * ldC + col] = v;
            }
        }
    }
}

// ---------------------------------------------------------------- BN stats (two-pass via atomics)
__global__ void k_zero(float* p, int n) {
    int i = blockIdx.x * 256 + threadIdx.x;
    if (i < n) p[i] = 0.f;
}
__global__ __launch_bounds__(256) void k_bn_partial(const float* __restrict__ X,
                                                    float* __restrict__ sum, float* __restrict__ sumsq) {
    int col = threadIdx.x;                  // HD=256
    int r0 = blockIdx.x * 128;              // 32 blocks x 128 rows
    float s = 0.f, q = 0.f;
    for (int row = r0; row < r0 + 128; ++row) {
        float v = X[(long)row * HD + col];
        s += v; q += v * v;
    }
    atomicAdd(&sum[col], s);
    atomicAdd(&sumsq[col], q);
}
__global__ void k_bn_final(const float* __restrict__ sum, const float* __restrict__ sumsq,
                           float* __restrict__ mu, float* __restrict__ rstd) {
    int c = threadIdx.x;
    float m = sum[c] / (float)NN;
    float v = sumsq[c] / (float)NN - m * m;
    mu[c] = m;
    rstd[c] = rsqrtf(v + BN_EPS);
}

// ---------------------------------------------------------------- fused BN + LN + ELU -> bf16 activations
__global__ __launch_bounds__(256) void k_bn_ln_elu(const float* __restrict__ X,
                                                   const float* __restrict__ mu, const float* __restrict__ rstd,
                                                   const float* __restrict__ bg, const float* __restrict__ bb,
                                                   const float* __restrict__ lg, const float* __restrict__ lb,
                                                   __bf16* __restrict__ out) {
    __shared__ float red[256];
    int row = blockIdx.x, c = threadIdx.x;
    float v = (X[(long)row * HD + c] - mu[c]) * rstd[c] * bg[c] + bb[c];
    // mean over HD
    red[c] = v; __syncthreads();
    for (int w = 128; w > 0; w >>= 1) { if (c < w) red[c] += red[c + w]; __syncthreads(); }
    float m = red[0] / (float)HD; __syncthreads();
    // var over HD
    float d = v - m;
    red[c] = d * d; __syncthreads();
    for (int w = 128; w > 0; w >>= 1) { if (c < w) red[c] += red[c + w]; __syncthreads(); }
    float var = red[0] / (float)HD;
    float y = d * rsqrtf(var + LN_EPS) * lg[c] + lb[c];
    y = (y > 0.f) ? y : (__expf(y) - 1.f);
    out[(long)row * HD + c] = (__bf16)y;
}

// ================================================================ workspace layout (bytes)
#define OFF_THETA  0L                                  // R*N*E bf16      = 67,108,864
#define OFF_ACT    (OFF_THETA  + 67108864L)            // N*HD bf16       =  2,097,152
#define OFF_T32    (OFF_ACT    + 2097152L)             // R*E*HD f32      =  8,388,608
#define OFF_TBF    (OFF_T32    + 8388608L)             // R*E*HD bf16     =  4,194,304
#define OFF_HN32   (OFF_TBF    + 4194304L)             // R*N*HD f32      = 16,777,216
#define OFF_HNBF   (OFF_HN32   + 16777216L)            // R*N*HD bf16     =  8,388,608
#define OFF_WBF    (OFF_HNBF   + 8388608L)             // R*HD*HD bf16    =    524,288
#define OFF_CONV   (OFF_WBF    + 524288L)              // N*HD f32        =  4,194,304
#define OFF_SN     (OFF_CONV   + 4194304L)             // R*N f32
#define OFF_SE     (OFF_SN     + 65536L)               // R*E f32
#define OFF_RW     (OFF_SE     + 32768L)               // R f32
#define OFF_BNSUM  (OFF_RW     + 256L)                 // 2*HD f32
#define OFF_BNST   (OFF_BNSUM  + 2048L)                // 2*HD f32

extern "C" void kernel_launch(void* const* d_in, const int* in_sizes, int n_in,
                              void* d_out, int out_size, void* d_ws, size_t ws_size,
                              hipStream_t stream) {
    const float* X        = (const float*)d_in[0];
    const float* H        = (const float*)d_in[1];
    const float* Wmat[3]  = { (const float*)d_in[2], (const float*)d_in[3], (const float*)d_in[4] };
    const float* imp      = (const float*)d_in[5];
    const float* bvec[3]  = { (const float*)d_in[6], (const float*)d_in[7], (const float*)d_in[8] };
    const float* bn_g     = (const float*)d_in[9];
    const float* bn_b     = (const float*)d_in[10];
    const float* ln_g     = (const float*)d_in[11];
    const float* ln_b     = (const float*)d_in[12];
    const float* rel_attn = (const float*)d_in[13];
    float* out = (float*)d_out;

    char* w = (char*)d_ws;
    __bf16* theta = (__bf16*)(w + OFF_THETA);
    __bf16* act   = (__bf16*)(w + OFF_ACT);
    float*  t32   = (float*) (w + OFF_T32);
    __bf16* tbf   = (__bf16*)(w + OFF_TBF);
    float*  hn32  = (float*) (w + OFF_HN32);
    __bf16* hnbf  = (__bf16*)(w + OFF_HNBF);
    __bf16* wbf   = (__bf16*)(w + OFF_WBF);
    float*  conv  = (float*) (w + OFF_CONV);
    float*  sN    = (float*) (w + OFF_SN);
    float*  sE    = (float*) (w + OFF_SE);
    float*  rw    = (float*) (w + OFF_RW);
    float*  bnsum = (float*) (w + OFF_BNSUM);
    float*  bnst  = (float*) (w + OFF_BNST);

    // --- theta materialization (one HBM sweep of H; theta bf16 stays in L2)
    k_prep  <<<1, 32, 0, stream>>>(rel_attn, rw);
    k_scaleN<<<dim3(NN, RR), 256, 0, stream>>>(H, rw, sN);
    k_scaleE<<<(RR * EE) / 256, 256, 0, stream>>>(H, rw, sE);
    k_theta <<<(unsigned)(((long)RR * NN * EE) / 256), 256, 0, stream>>>(H, rw, sN, sE, theta);

    // --- X -> bf16 activations
    k_cvt<<<(NN * FF) / 256, 256, 0, stream>>>(X, act);

    for (int l = 0; l < 3; ++l) {
        int Fin  = (l == 0) ? FF : HD;
        int Fout = (l == 2) ? FF : HD;

        // t[r] = theta[r]^T @ Xin    [E x Fin], K = NN
        gemm_bf16<true, false><<<dim3(Fin / 64, EE / 128, RR), 256, 0, stream>>>(
            theta, act, t32, EE, Fin, NN, EE, Fin, Fin,
            (long)NN * EE, 0L, (long)EE * Fin, RR, nullptr, nullptr);
        k_cvt<<<(RR * EE * Fin) / 256, 256, 0, stream>>>(t32, tbf);

        // Hn[r] = theta[r] @ t[r]    [N x Fin], K = EE
        gemm_bf16<false, false><<<dim3(Fin / 64, NN / 128, RR), 256, 0, stream>>>(
            theta, tbf, hn32, NN, Fin, EE, EE, Fin, Fin,
            (long)NN * EE, (long)EE * Fin, (long)NN * Fin, RR, nullptr, nullptr);
        k_cvt<<<(RR * NN * Fin) / 256, 256, 0, stream>>>(hn32, hnbf);

        // Wb[r] = sigmoid(imp[l,r]) * W[r]   (fold relation importance into weights)
        k_wconv<<<(RR * Fin * Fout) / 256, 256, 0, stream>>>(Wmat[l], imp, l, Fin * Fout, wbf);

        // out = sum_r Hn[r] @ Wb[r] + bias (+X residual on last layer)
        float* Cdst = (l == 2) ? out : conv;
        const float* resid = (l == 2) ? X : nullptr;
        gemm_bf16<false, true><<<dim3(Fout / 64, NN / 128, 1), 256, 0, stream>>>(
            hnbf, wbf, Cdst, NN, Fout, Fin, Fin, Fout, Fout,
            (long)NN * Fin, (long)Fin * Fout, 0L, RR, bvec[l], resid);

        if (l < 2) {
            k_zero<<<2, 256, 0, stream>>>(bnsum, 2 * HD);
            k_bn_partial<<<NN / 128, 256, 0, stream>>>(conv, bnsum, bnsum + HD);
            k_bn_final<<<1, HD, 0, stream>>>(bnsum, bnsum + HD, bnst, bnst + HD);
            k_bn_ln_elu<<<NN, HD, 0, stream>>>(conv, bnst, bnst + HD,
                                               bn_g + l * HD, bn_b + l * HD,
                                               ln_g + l * HD, ln_b + l * HD, act);
        }
    }
}
